// UCBAttention_73418170958448
// MI455X (gfx1250) — matrix operand order repaired and checked by reading, not compiled, not run on hardware
//
// UCB-Attention for MI455X (gfx1250, wave32, WMMA).
// All four matmuls (QKV, Q*K^T, P*V, proj) run on v_wmma_f32_16x16x32_bf16.
// Fragment loads are unguarded 16B vector loads (M-dims padded to x16 and
// seq dim padded to NR=608, pads zeroed per call). V is stored transposed
// so every WMMA B-operand load is two contiguous 16B runs.
// Workspace requirement: ~341 MB (S f32 is the big one).
#include <hip/hip_runtime.h>
#include <hip/hip_bf16.h>
#include <math.h>
#include <stdint.h>

#define Bb     8
#define Nq     577          // tokens
#define Cdim   1024
#define Hh     16
#define Dd     64
#define Mrows  (Bb * Nq)    // 4616
#define Mpad   4624         // 289*16
#define NR     608          // padded seq (19*32, covers 577)
#define KKEEP  288
#define SCL    0.125f       // D^-0.5
#define BETA   1.0f

typedef __bf16 bf16;
typedef __attribute__((ext_vector_type(16))) __bf16 v16bf;
typedef __attribute__((ext_vector_type(8)))  __bf16 v8bf;
typedef __attribute__((ext_vector_type(8)))  float  v8f;

static __device__ __forceinline__ v8f wmma_bf16(v16bf a, v16bf b, v8f c) {
  // D = A(16x32 bf16) * B(32x16 bf16) + C(16x16 f32)
  return __builtin_amdgcn_wmma_f32_16x16x32_bf16(false, a, false, b, (short)0, c,
                                                 false, false);
}
static __device__ __forceinline__ v16bf cat8(v8bf lo, v8bf hi) {
  return __builtin_shufflevector(lo, hi, 0, 1, 2, 3, 4, 5, 6, 7,
                                 8, 9, 10, 11, 12, 13, 14, 15);
}

// ---- fragment loaders (ISA 7.12.2 layouts, wave32), unguarded --------------
// A 16x32 bf16: lane holds row m=lane&15; per-lane data = two 16B chunks at
// k0+hf*8 and k0+16+hf*8.  Caller guarantees rows/pads are in-bounds & zeroed.
static __device__ __forceinline__ v16bf load_a(const bf16* base, int lda,
                                               int row0, int k0) {
  int l = threadIdx.x & 31, m = l & 15, hf = l >> 4;
  const bf16* p = base + (size_t)(row0 + m) * lda + k0 + hf * 8;
  v8bf lo = *(const v8bf*)(p);
  v8bf hi = *(const v8bf*)(p + 16);
  return cat8(lo, hi);
}
// B 32x16 with B[k][n] = W[n][k] (W row-major): lane holds col n, two
// 16B chunks of K at k0+hf*8 and k0+16+hf*8.
static __device__ __forceinline__ v16bf load_bT(const bf16* W, int ldw,
                                                int n0, int k0) {
  int l = threadIdx.x & 31, n = l & 15, hf = l >> 4;
  const bf16* p = W + (size_t)(n0 + n) * ldw + k0 + hf * 8;
  v8bf lo = *(const v8bf*)(p);
  v8bf hi = *(const v8bf*)(p + 16);
  return cat8(lo, hi);
}

// ---- utility kernels -------------------------------------------------------
__global__ void zero_u32_kernel(uint32_t* p, size_t n) {
  size_t i = (size_t)blockIdx.x * blockDim.x + threadIdx.x;
  size_t st = (size_t)gridDim.x * blockDim.x;
  for (; i < n; i += st) p[i] = 0u;
}
__global__ void f32_to_bf16_kernel(const float* s, bf16* d, size_t n) {
  size_t i = (size_t)blockIdx.x * blockDim.x + threadIdx.x;
  size_t st = (size_t)gridDim.x * blockDim.x;
  for (; i < n; i += st) d[i] = (bf16)s[i];
}

// ---- K1: qkv = x @ Wqkv^T + b; q,k -> [B,H,NR,D], v -> [B,H,D,NR] (transp) -
__global__ void qkv_gemm_kernel(const bf16* __restrict__ xb,
                                const bf16* __restrict__ wb,
                                const float* __restrict__ bias,
                                bf16* qo, bf16* ko, bf16* vt) {
  int wave = threadIdx.x >> 5;
  int rt = blockIdx.x;                 // 0..288 (row tiles of 16)
  int ct = blockIdx.y * 8 + wave;      // 0..191 (col tiles of 16, 3072 cols)
  int row0 = rt * 16, col0 = ct * 16;
  v8f acc = {};
  for (int k0 = 0; k0 < Cdim; k0 += 32) {
    __builtin_prefetch(xb + (size_t)row0 * Cdim + k0 + 128, 0, 3);
    v16bf a = load_a(xb, Cdim, row0, k0);
    v16bf b = load_bT(wb, Cdim, col0, k0);
    acc = wmma_bf16(a, b, acc);
  }
  int l = threadIdx.x & 31, n = l & 15, hf = l >> 4;
  int col = col0 + n;
  int which = col >> 10, rem = col & 1023, h = rem >> 6, d = rem & 63;
  float bv = bias[col];
#pragma unroll
  for (int r = 0; r < 8; ++r) {
    int row = row0 + r + 8 * hf;
    if (row < Mrows) {
      int b_ = row / Nq, n_ = row - b_ * Nq;
      bf16 val = (bf16)(acc[r] + bv);
      if (which == 0)
        qo[(((size_t)b_ * Hh + h) * NR + n_) * Dd + d] = val;
      else if (which == 1)
        ko[(((size_t)b_ * Hh + h) * NR + n_) * Dd + d] = val;
      else
        vt[(((size_t)b_ * Hh + h) * Dd + d) * NR + n_] = val;   // transposed
    }
  }
}

// ---- K2: S = scale * q @ k^T, f32 [B*H, NR, NR] (rows/cols 0..591 written) -
__global__ void attn_scores_kernel(const bf16* __restrict__ qb,
                                   const bf16* __restrict__ kb,
                                   float* __restrict__ S) {
  int bh = blockIdx.z;
  int rt = blockIdx.x;                              // 0..36
  int ct = blockIdx.y * 4 + (threadIdx.x >> 5);     // 0..39
  if (ct >= 37) return;
  const bf16* qbase = qb + (size_t)bh * NR * Dd;
  const bf16* kbase = kb + (size_t)bh * NR * Dd;
  v8f acc = {};
#pragma unroll
  for (int k0 = 0; k0 < Dd; k0 += 32) {
    v16bf a = load_a(qbase, Dd, rt * 16, k0);       // pads are zero
    v16bf b = load_bT(kbase, Dd, ct * 16, k0);
    acc = wmma_bf16(a, b, acc);
  }
  int l = threadIdx.x & 31, n = l & 15, hf = l >> 4;
  float* Sb = S + (size_t)bh * NR * NR;
  int col = ct * 16 + n;
#pragma unroll
  for (int r = 0; r < 8; ++r) {
    int row = rt * 16 + r + 8 * hf;
    Sb[(size_t)row * NR + col] = acc[r] * SCL;
  }
}

// ---- K3: row softmax -> P bf16; accumulate column sums (patch scores) ------
__global__ void softmax_kernel(const float* __restrict__ S,
                               bf16* __restrict__ P,
                               float* __restrict__ patch) {
  int gw = blockIdx.x * (blockDim.x >> 5) + (threadIdx.x >> 5);
  const int total = Bb * Hh * 592;
  if (gw >= total) return;
  int bh = gw / 592, row = gw - bh * 592;
  const float* s = S + ((size_t)bh * NR + row) * NR;
  bf16* p = P + ((size_t)bh * NR + row) * NR;
  int l = threadIdx.x & 31;
  float mx = -1e30f;
  for (int c = l; c < Nq; c += 32) mx = fmaxf(mx, s[c]);
#pragma unroll
  for (int o = 16; o; o >>= 1) mx = fmaxf(mx, __shfl_xor(mx, o, 32));
  float sum = 0.0f;
  for (int c = l; c < Nq; c += 32) sum += __expf(s[c] - mx);
#pragma unroll
  for (int o = 16; o; o >>= 1) sum += __shfl_xor(sum, o, 32);
  float inv = 1.0f / sum;
  bool valid_q = (row < Nq);
  float* pr = patch + (size_t)bh * Nq;
  for (int c = l; c < NR; c += 32) {
    float pv = (c < Nq) ? __expf(s[c] - mx) * inv : 0.0f;
    p[c] = (bf16)pv;
    if (valid_q && c >= 1 && c < Nq) atomicAdd(&pr[c], pv);
  }
}

// ---- K4: UCB scores + exact top-288 (rank method) per batch ----------------
__global__ void ucb_topk_kernel(const float* __restrict__ patch,
                                const float* __restrict__ ucnt,
                                const int* __restrict__ counter,
                                float* __restrict__ keep,
                                float* __restrict__ score_delta) {
  __shared__ float g[Nq];
  int b = blockIdx.x;
  float logc = logf((float)(*counter) + 1.0f);
  for (int col = 1 + threadIdx.x; col < Nq; col += blockDim.x) {
    float acc = 0.0f;
    for (int h = 0; h < Hh; ++h) {
      float pm = patch[((size_t)b * Hh + h) * Nq + col] * (1.0f / (float)Nq);
      float ex = sqrtf(logc / (ucnt[h * Nq + col] + 1e-6f));
      acc += pm + BETA * ex;
    }
    g[col] = acc * (1.0f / (float)Hh);
  }
  __syncthreads();
  if (threadIdx.x == 0) keep[(size_t)b * NR + 0] = 1.0f;   // CLS always kept
  for (int col = 1 + threadIdx.x; col < Nq; col += blockDim.x) {
    float gv = g[col];
    int rank = 0;
    for (int c = 1; c < Nq; ++c) {
      float o = g[c];
      rank += (o > gv) || (o == gv && c < col);   // top_k tie-break: low index
    }
    float kf = (rank < KKEEP) ? 1.0f : 0.0f;
    keep[(size_t)b * NR + col] = kf;
    if (kf > 0.0f)
      for (int h = 0; h < Hh; ++h)
        atomicAdd(&score_delta[h * Nq + col], 1.0f / (float)Bb);
  }
  for (int col = Nq + threadIdx.x; col < NR; col += blockDim.x)
    keep[(size_t)b * NR + col] = 0.0f;
}

// ---- K5: masked/renormalized context = (P .* mask)/den @ V ------------------
// V is pre-transposed: vt[B,H,Dd,NR], so B-fragments are two contiguous 16B runs per lane.
__global__ void context_kernel(const bf16* __restrict__ P,
                               const bf16* __restrict__ vt,
                               const float* __restrict__ keep,
                               bf16* __restrict__ ctx) {
  __shared__ float keepc[NR];
  int bh = blockIdx.y, b = bh / Hh, h = bh - b * Hh;
  int rt = blockIdx.x;                       // 0..36
  for (int c = threadIdx.x; c < NR; c += 32) keepc[c] = keep[(size_t)b * NR + c];
  __syncthreads();
  const bf16* Pb = P + (size_t)bh * NR * NR;
  const bf16* Vt = vt + (size_t)bh * Dd * NR;
  int l = threadIdx.x & 31, m = l & 15, hf = l >> 4;
  int row = rt * 16 + m;
  const bf16* prow = Pb + (size_t)row * NR;
  bool rk = (row >= Nq) || (keepc[row] > 0.0f);  // kept row -> full mask
  float rowm = rk ? 1.0f : 0.0f;
  // per-row renorm denominator (split cols across wave halves)
  float den = 0.0f;
  for (int c = hf; c < Nq; c += 2) {
    float pv = (float)prow[c];
    den += pv * fmaxf(rowm, keepc[c]);
  }
  den += __shfl_xor(den, 16, 32);
  float rinv = 1.0f / (den + 1e-8f);          // lane m holds 1/den of row m
  v8f a0 = {}, a1 = {}, a2 = {}, a3 = {};
  for (int c0 = 0; c0 < NR; c0 += 32) {
    // masked A fragment: two 16B chunks, branch-free mask multiply
    const bf16* pc = prow + c0 + hf * 8;
    v8bf lo = *(const v8bf*)(pc);
    v8bf hi = *(const v8bf*)(pc + 16);
    v16bf af;
#pragma unroll
    for (int e = 0; e < 8; ++e) {
      int klo = c0 + hf * 8 + e;
      int khi = klo + 16;
      af[e]     = (bf16)((float)lo[e] * fmaxf(rowm, keepc[klo]));
      af[e + 8] = (bf16)((float)hi[e] * fmaxf(rowm, keepc[khi]));
    }
    a0 = wmma_bf16(af, load_bT(Vt, NR, 0,  c0), a0);
    a1 = wmma_bf16(af, load_bT(Vt, NR, 16, c0), a1);
    a2 = wmma_bf16(af, load_bT(Vt, NR, 32, c0), a2);
    a3 = wmma_bf16(af, load_bT(Vt, NR, 48, c0), a3);
  }
  int n = l & 15;
#pragma unroll
  for (int r = 0; r < 8; ++r) {
    int qrow = rt * 16 + r + 8 * hf;
    if (qrow >= Nq) continue;
    float ri = __shfl(rinv, r + 8 * hf, 32);
    bf16* o = ctx + ((size_t)b * Nq + qrow) * Cdim + h * Dd;
    o[0 * 16 + n] = (bf16)(a0[r] * ri);
    o[1 * 16 + n] = (bf16)(a1[r] * ri);
    o[2 * 16 + n] = (bf16)(a2[r] * ri);
    o[3 * 16 + n] = (bf16)(a3[r] * ri);
  }
}

// ---- K6: out = ctx @ Wproj^T + bproj ---------------------------------------
__global__ void proj_gemm_kernel(const bf16* __restrict__ ctx,
                                 const bf16* __restrict__ wpb,
                                 const float* __restrict__ bias,
                                 float* __restrict__ out) {
  int rt = blockIdx.x;                              // 0..288
  int ct = blockIdx.y * 4 + (threadIdx.x >> 5);     // 0..63
  int row0 = rt * 16, col0 = ct * 16;
  v8f acc = {};
  for (int k0 = 0; k0 < Cdim; k0 += 32) {
    __builtin_prefetch(ctx + (size_t)row0 * Cdim + k0 + 128, 0, 3);
    v16bf a = load_a(ctx, Cdim, row0, k0);
    v16bf b = load_bT(wpb, Cdim, col0, k0);
    acc = wmma_bf16(a, b, acc);
  }
  int l = threadIdx.x & 31, n = l & 15, hf = l >> 4;
  int col = col0 + n;
  float bv = bias[col];
#pragma unroll
  for (int r = 0; r < 8; ++r) {
    int row = row0 + r + 8 * hf;
    if (row < Mrows) out[(size_t)row * Cdim + col] = acc[r] + bv;
  }
}

extern "C" void kernel_launch(void* const* d_in, const int* in_sizes, int n_in,
                              void* d_out, int out_size, void* d_ws, size_t ws_size,
                              hipStream_t stream) {
  const float* x     = (const float*)d_in[0];
  const float* Wqkv  = (const float*)d_in[1];
  const float* bqkv  = (const float*)d_in[2];
  const float* Wproj = (const float*)d_in[3];
  const float* bproj = (const float*)d_in[4];
  const float* ucnt  = (const float*)d_in[5];
  const int*   cnt   = (const int*)d_in[6];
  float* out = (float*)d_out;
  float* sdelta = out + (size_t)Mrows * Cdim;      // H*Nq tail of d_out

  // workspace carve-out (~341 MB total)
  char* ws = (char*)d_ws;
  size_t off = 0;
  auto carve = [&](size_t bytes) -> void* {
    void* p = ws + off;
    off = (off + bytes + 255) & ~(size_t)255;
    return p;
  };
  bf16*  xb    = (bf16*)carve((size_t)Mpad * Cdim * 2);   // M padded to x16
  bf16*  wqb   = (bf16*)carve((size_t)3 * Cdim * Cdim * 2);
  bf16*  wpb   = (bf16*)carve((size_t)Cdim * Cdim * 2);
  bf16*  qb    = (bf16*)carve((size_t)Bb * Hh * NR * Dd * 2);
  bf16*  kb    = (bf16*)carve((size_t)Bb * Hh * NR * Dd * 2);
  bf16*  vt    = (bf16*)carve((size_t)Bb * Hh * Dd * NR * 2);  // transposed V
  float* patch = (float*)carve((size_t)Bb * Hh * Nq * 4);
  float* keep  = (float*)carve((size_t)Bb * NR * 4);
  float* S     = (float*)carve((size_t)Bb * Hh * NR * NR * 4);
  bf16*  P     = (bf16*)carve((size_t)Bb * Hh * NR * NR * 2);
  bf16*  ctx   = (bf16*)carve((size_t)Mpad * Cdim * 2);   // M padded to x16

  // 0) zero accumulators / padded tensors (every call: harness never re-poisons)
  size_t qkvU32 = (size_t)Bb * Hh * NR * Dd * 2 / 4;
  size_t padU32 = (size_t)(Mpad - Mrows) * Cdim * 2 / 4;  // 4096
  zero_u32_kernel<<<1024, 256, 0, stream>>>((uint32_t*)qb, qkvU32);
  zero_u32_kernel<<<1024, 256, 0, stream>>>((uint32_t*)kb, qkvU32);
  zero_u32_kernel<<<1024, 256, 0, stream>>>((uint32_t*)vt, qkvU32);
  zero_u32_kernel<<<16, 256, 0, stream>>>(
      (uint32_t*)(xb + (size_t)Mrows * Cdim), padU32);
  zero_u32_kernel<<<16, 256, 0, stream>>>(
      (uint32_t*)(ctx + (size_t)Mrows * Cdim), padU32);
  zero_u32_kernel<<<256, 256, 0, stream>>>((uint32_t*)patch, (size_t)Bb * Hh * Nq);
  zero_u32_kernel<<<8, 256, 0, stream>>>((uint32_t*)keep, (size_t)Bb * NR);
  zero_u32_kernel<<<8, 256, 0, stream>>>((uint32_t*)sdelta, (size_t)Hh * Nq);

  // 1) f32 -> bf16 operands
  f32_to_bf16_kernel<<<2048, 256, 0, stream>>>(x, xb, (size_t)Mrows * Cdim);
  f32_to_bf16_kernel<<<2048, 256, 0, stream>>>(Wqkv, wqb, (size_t)3 * Cdim * Cdim);
  f32_to_bf16_kernel<<<1024, 256, 0, stream>>>(Wproj, wpb, (size_t)Cdim * Cdim);

  // 2) QKV GEMM (WMMA)
  qkv_gemm_kernel<<<dim3(289, 24), 256, 0, stream>>>(xb, wqb, bqkv, qb, kb, vt);
  // 3) attention scores (WMMA)
  attn_scores_kernel<<<dim3(37, 10, Bb * Hh), 128, 0, stream>>>(qb, kb, S);
  // 4) softmax + patch column sums
  {
    int waves = Bb * Hh * 592;
    softmax_kernel<<<(waves + 7) / 8, 256, 0, stream>>>(S, P, patch);
  }
  // 5) UCB + top-K selection + score_delta
  ucb_topk_kernel<<<Bb, 256, 0, stream>>>(patch, ucnt, cnt, keep, sdelta);
  // 6) masked renormalized context (WMMA)
  context_kernel<<<dim3(37, Bb * Hh), 32, 0, stream>>>(P, vt, keep, ctx);
  // 7) output projection (WMMA)
  proj_gemm_kernel<<<dim3(289, 16), 128, 0, stream>>>(ctx, wpb, bproj, out);
}
